// StandardAttentionLayer_5927054868632
// MI455X (gfx1250) — matrix operand order repaired and checked by reading, not compiled, not run on hardware
//
#include <hip/hip_runtime.h>

typedef __attribute__((ext_vector_type(16))) _Float16 v16h;
typedef __attribute__((ext_vector_type(8)))  _Float16 v8h;
typedef __attribute__((ext_vector_type(4)))  _Float16 v4h;
typedef __attribute__((ext_vector_type(2)))  _Float16 v2h;
typedef __attribute__((ext_vector_type(8)))  float    v8f;
typedef __attribute__((ext_vector_type(4)))  float    v4f;
typedef __attribute__((ext_vector_type(4)))  unsigned u32x4;
typedef __attribute__((ext_vector_type(8)))  int      i32x8;
typedef __attribute__((ext_vector_type(4)))  int      i32x4;

static constexpr int Bc = 2, Hc = 16, Sc = 2048, Dc = 64;
static constexpr int BLOCK_M = 64;   // Q rows per workgroup (4 waves x 16)
static constexpr int BLOCK_N = 64;   // KV rows per iteration
static constexpr int NBH     = Bc * Hc;
static constexpr float QSCALE = 0.125f;   // 1/sqrt(64)

__device__ __forceinline__ v8f wmma_f16(v16h a, v16h b, v8f c) {
    return __builtin_amdgcn_wmma_f32_16x16x32_f16(false, a, false, b,
                                                  (short)0, c, false, false);
}
// v_cvt_pk_rtz_f16_f32 wrapper (builtin returns __fp16v2; bit_cast to our v2h)
__device__ __forceinline__ v2h cvt_pk(float a, float b) {
    return __builtin_bit_cast(v2h, __builtin_amdgcn_cvt_pkrtz(a, b));
}
__device__ __forceinline__ unsigned cvt_pk_u32(float a, float b) {
    return __builtin_bit_cast(unsigned, __builtin_amdgcn_cvt_pkrtz(a, b));
}
// Exchange with lane^16 (cross-half identity permute) -- pure VALU, no LDS.
__device__ __forceinline__ unsigned xhalf_u32(unsigned u) {
    return __builtin_amdgcn_permlanex16(u, u, 0x76543210u, 0xFEDCBA98u, false, false);
}
__device__ __forceinline__ float xhalf_f32(float x) {
    return __builtin_bit_cast(float, xhalf_u32(__builtin_bit_cast(unsigned, x)));
}
// A-fragment load (contiguous k order): e -> K=hf*16+e
__device__ __forceinline__ v16h loadA(const _Float16* p, int hf) {
    v8h lo = *(const v8h*)(p + hf * 16);
    v8h hi = *(const v8h*)(p + hf * 16 + 8);
    v16h a;
#pragma unroll
    for (int e = 0; e < 8; ++e) { a[e] = lo[e]; a[8 + e] = hi[e]; }
    return a;
}

// ---- TDM: issue one 2D f16 tile load Global -> LDS (D# per ISA 08 §8) -------
// tile_d0 elements per row (contiguous), tile_d1 rows, row pitch stride0 elems.
// LDS receives rows packed contiguously (tile_d0*2 bytes per row).
__device__ __forceinline__ void tdm_load_f16_2d(unsigned lds_addr, const void* gaddr,
                                                unsigned tile_d0, unsigned tile_d1,
                                                unsigned stride0) {
    const unsigned long long ga = (unsigned long long)gaddr;
    u32x4 g0;
    g0.x = 1u;                                          // count=1 (valid user D#)
    g0.y = lds_addr;                                    // LDS byte address
    g0.z = (unsigned)(ga & 0xFFFFFFFFu);                // global_addr[31:0]
    g0.w = (unsigned)((ga >> 32) & 0x01FFFFFFu)         // global_addr[56:32]
         | 0x80000000u;                                 // type=2 ("image")
    i32x8 g1;
    g1[0] = 0x00010000;                                 // wg_mask=0, data_size=1 (2B)
    g1[1] = (int)(tile_d0 << 16);                       // tensor_dim0[15:0] = tile_d0
    g1[2] = (int)((tile_d0 >> 16) | (tile_d1 << 16));   // tensor_dim0[31:16]|tensor_dim1[15:0]
    g1[3] = (int)((tile_d1 >> 16) | (tile_d0 << 16));   // tensor_dim1[31:16]|tile_dim0
    g1[4] = (int)(tile_d1 & 0xFFFFu);                   // tile_dim1 (tile_dim2=0)
    g1[5] = (int)stride0;                               // tensor_dim0_stride[31:0]
    g1[6] = 0;                                          // stride0[47:32], dim1_stride lo
    g1[7] = 0;
    const i32x4 z4 = {0, 0, 0, 0};                      // groups 2/3: <=2D tensor
    const i32x8 z8 = {0, 0, 0, 0, 0, 0, 0, 0};
    __builtin_amdgcn_tensor_load_to_lds(g0, g1, z4, z4, z8, 0);
}
__device__ __forceinline__ unsigned lds_addr_of(const void* p) {
    return (unsigned)(unsigned long long)p;             // LDS aperture: addr[31:0]
}

// ---- one 64-KV block of flash attention (shared by both paths) --------------
__device__ __forceinline__ void fa_block(const _Float16* __restrict__ Klc,
                                         const _Float16* __restrict__ VTlc,
                                         const _Float16* __restrict__ kplc,
                                         int l16, int hf, const v16h* bQ,
                                         v8f* o, v8f& lsum, float& mst,
                                         const v8f& zro)
{
    // S^T = K x Q^T : 4 kv-tiles, K-dim d=64 (2 chained WMMAs each)
    v8f s[4];
#pragma unroll
    for (int nt = 0; nt < 4; ++nt) {
        const _Float16* kr = Klc + (nt * 16 + l16) * 64;
        s[nt] = wmma_f16(loadA(kr,      hf), bQ[0], zro);
        s[nt] = wmma_f16(loadA(kr + 32, hf), bQ[1], s[nt]);
    }
    // online softmax: KV lies in-lane; single cross-half exchange for the max
    float mx = s[0][0];
#pragma unroll
    for (int nt = 0; nt < 4; ++nt)
#pragma unroll
        for (int r = 0; r < 8; ++r) mx = fmaxf(mx, s[nt][r]);
    mx = fmaxf(mx, xhalf_f32(mx));
    const float mnew = fmaxf(mst, mx);
    const float rs   = __expf(mst - mnew);
    mst = mnew;

    float p[4][8];
#pragma unroll
    for (int nt = 0; nt < 4; ++nt)
#pragma unroll
        for (int r = 0; r < 8; ++r) p[nt][r] = __expf(s[nt][r] - mnew);
#pragma unroll
    for (int t = 0; t < 4; ++t)
#pragma unroll
        for (int r = 0; r < 8; ++r) o[t][r] *= rs;
#pragma unroll
    for (int r = 0; r < 8; ++r) lsum[r] *= rs;

    // P^T C-layout -> B-fragments via packed cross-half exchange (register-only)
    v16h bP[2];
#pragma unroll
    for (int c = 0; c < 2; ++c) {
        unsigned ownp[4], sndp[4], rcv[4];
#pragma unroll
        for (int j = 0; j < 4; ++j) {
            const float o0 = hf ? p[2*c+1][2*j]   : p[2*c][2*j];
            const float o1 = hf ? p[2*c+1][2*j+1] : p[2*c][2*j+1];
            const float s0 = hf ? p[2*c][2*j]     : p[2*c+1][2*j];
            const float s1 = hf ? p[2*c][2*j+1]   : p[2*c+1][2*j+1];
            ownp[j] = cvt_pk_u32(o0, o1);
            sndp[j] = cvt_pk_u32(s0, s1);
        }
#pragma unroll
        for (int j = 0; j < 4; ++j) rcv[j] = xhalf_u32(sndp[j]);
        union { unsigned u[8]; v16h v; } bld;
#pragma unroll
        for (int j = 0; j < 4; ++j) {
            bld.u[j]     = hf ? rcv[j]  : ownp[j];
            bld.u[4 + j] = hf ? ownp[j] : rcv[j];
        }
        bP[c] = bld.v;
    }
    // lsum += keep x P^T (softmax denominator via WMMA; mask folded in)
#pragma unroll
    for (int c = 0; c < 2; ++c)
        lsum = wmma_f16(loadA(kplc + c * 32, hf), bP[c], lsum);
    // O^T += V^T x P^T : 4 d-tiles x 2 kv-chunks
#pragma unroll
    for (int t = 0; t < 4; ++t)
#pragma unroll
        for (int c = 0; c < 2; ++c)
            o[t] = wmma_f16(loadA(VTlc + (t * 16 + l16) * 64 + c * 32, hf), bP[c], o[t]);
}

__device__ __forceinline__ void fa_epilogue(float* __restrict__ Ob, int qr, int hf,
                                            const v8f* o, const v8f& lsum)
{
    const float inv = 1.0f / lsum[0];
    float* op = Ob + (size_t)qr * Dc;
#pragma unroll
    for (int t = 0; t < 4; ++t) {
        v4f f0, f1;
#pragma unroll
        for (int r = 0; r < 4; ++r) { f0[r] = o[t][r] * inv; f1[r] = o[t][4+r] * inv; }
        *(v4f*)(op + t * 16 + 8 * hf)     = f0;
        *(v4f*)(op + t * 16 + 8 * hf + 4) = f1;
    }
}

// ---- pre-pass: K -> f16 (flat convert) --------------------------------------
__global__ __launch_bounds__(256)
void conv_k_kernel(const float* __restrict__ K, _Float16* __restrict__ Kh) {
    const size_t idx = ((size_t)blockIdx.x * 256 + threadIdx.x) * 8;
    v8h h;
#pragma unroll
    for (int j = 0; j < 4; ++j) {
        v2h pk = cvt_pk(K[idx + 2*j], K[idx + 2*j + 1]);
        h[2*j] = pk[0]; h[2*j+1] = pk[1];
    }
    *(v8h*)(Kh + idx) = h;
}

// ---- pre-pass: V -> f16, transposed per (b,h), mask-zeroed; also keep[] -----
__global__ __launch_bounds__(256)
void conv_vt_kernel(const float* __restrict__ V, const int* __restrict__ Msk,
                    _Float16* __restrict__ VTh, _Float16* __restrict__ keepH) {
    const int t   = blockIdx.x * 256 + threadIdx.x;   // 32*64*2048/8 threads
    const int bh  = t >> 14;            // /(64*256)
    const int rem = t & 16383;
    const int d   = rem >> 8;
    const int kv0 = (rem & 255) * 8;
    const int b   = bh >> 4;
    const float* vp = V + (size_t)bh * Sc * Dc;
    const int*   mp = Msk + (size_t)b * Sc;
    float f[8];
#pragma unroll
    for (int i = 0; i < 8; ++i)
        f[i] = mp[kv0 + i] ? vp[(size_t)(kv0 + i) * Dc + d] : 0.0f;
    v8h h;
#pragma unroll
    for (int j = 0; j < 4; ++j) {
        v2h pk = cvt_pk(f[2*j], f[2*j+1]);
        h[2*j] = pk[0]; h[2*j+1] = pk[1];
    }
    *(v8h*)(VTh + (size_t)bh * Dc * Sc + (size_t)d * Sc + kv0) = h;
    if (d == 0 && (bh & (Hc - 1)) == 0) {
        v8h kh;
#pragma unroll
        for (int i = 0; i < 8; ++i) kh[i] = (_Float16)(mp[kv0 + i] ? 1.0f : 0.0f);
        *(v8h*)(keepH + (size_t)b * Sc + kv0) = kh;
    }
}

// ---- main kernel, pre-converted path: TDM double-buffered staging -----------
__global__ __launch_bounds__(128, 1)
void fa_fwd_pre(const float* __restrict__ Q, const _Float16* __restrict__ Kh,
                const _Float16* __restrict__ VTh, const _Float16* __restrict__ keepH,
                float* __restrict__ O)
{
    __shared__ alignas(16) _Float16 Kl[2][BLOCK_N * Dc];
    __shared__ alignas(16) _Float16 VTl[2][Dc * BLOCK_N];
    __shared__ alignas(16) _Float16 kpL[2][BLOCK_N];

    const int tid  = threadIdx.x;
    const int lane = tid & 31;
    const int wave = tid >> 5;
    const int hf   = lane >> 4;
    const int l16  = lane & 15;

    const int nQblk = Sc / BLOCK_M;
    const int bh    = blockIdx.x / nQblk;
    const int qblk  = blockIdx.x % nQblk;
    const int b     = bh / Hc;

    const float*    Qb  = Q    + (size_t)bh * Sc * Dc;
    const _Float16* Khb = Kh   + (size_t)bh * Sc * Dc;
    const _Float16* VTb = VTh  + (size_t)bh * Dc * Sc;
    const _Float16* kpb = keepH + (size_t)b * Sc;
    float*          Ob  = O    + (size_t)bh * Sc * Dc;

    // Q^T B-fragments: lane = query column; element e -> d = c*32 + hf*16 + e
    const int qr = qblk * BLOCK_M + wave * 16 + l16;
    v16h bQ[2];
    {
        const float* qp = Qb + (size_t)qr * Dc;
#pragma unroll
        for (int c = 0; c < 2; ++c)
#pragma unroll
            for (int e = 0; e < 16; ++e)
                bQ[c][e] = (_Float16)(qp[c * 32 + hf * 16 + e] * QSCALE);
    }

    v8f o[4], lsum, zro;
#pragma unroll
    for (int r = 0; r < 8; ++r) { lsum[r] = 0.0f; zro[r] = 0.0f; }
#pragma unroll
    for (int t = 0; t < 4; ++t)
#pragma unroll
        for (int r = 0; r < 8; ++r) o[t][r] = 0.0f;
    float mst = -INFINITY;

    // One wave drives the Tensor Data Mover for all staging (EXEC-independent,
    // descriptor in SGPRs; tracked by TENSORcnt). Tiles:
    //   K  : flat 64*64 f16 (1 row of 4096), VT: 2D 64 rows x 64 f16 (pitch Sc),
    //   keep: 64 f16.
    auto stage = [&](int buf, int kb) {
        tdm_load_f16_2d(lds_addr_of(&Kl[buf][0]),  Khb + (size_t)kb * Dc,
                        BLOCK_N * Dc, 1, BLOCK_N * Dc);
        tdm_load_f16_2d(lds_addr_of(&VTl[buf][0]), VTb + kb,
                        BLOCK_N, Dc, Sc);
        tdm_load_f16_2d(lds_addr_of(&kpL[buf][0]), kpb + kb,
                        BLOCK_N, 1, BLOCK_N);
    };

    if (wave == 0) {
        stage(0, 0);
        __builtin_amdgcn_s_wait_tensorcnt(0);
    }
    __syncthreads();

    const int nblk = Sc / BLOCK_N;
    for (int i = 0; i < nblk; ++i) {
        const int  cur  = i & 1;
        const bool more = (i + 1) < nblk;
        // kick off next block's DMA before computing; overlaps with the WMMAs
        if (more && wave == 0) stage(cur ^ 1, (i + 1) * BLOCK_N);

        fa_block(Kl[cur], VTl[cur], kpL[cur], l16, hf, bQ, o, lsum, mst, zro);

        if (more && wave == 0) __builtin_amdgcn_s_wait_tensorcnt(0);
        __syncthreads();
    }

    fa_epilogue(Ob, qr, hf, o, lsum);
}

// ---- fallback path: in-kernel staging (no workspace needed) -----------------
__global__ __launch_bounds__(128, 1)
void fa_fwd_kernel(const float* __restrict__ Q, const float* __restrict__ K,
                   const float* __restrict__ V, const int* __restrict__ Msk,
                   float* __restrict__ O)
{
    __shared__ alignas(16) _Float16 Kl[BLOCK_N * Dc];
    __shared__ alignas(16) _Float16 VTl[Dc * BLOCK_N];
    __shared__ alignas(16) _Float16 keepL[BLOCK_N];

    const int tid  = threadIdx.x;
    const int lane = tid & 31;
    const int wave = tid >> 5;
    const int hf   = lane >> 4;
    const int l16  = lane & 15;

    const int nQblk = Sc / BLOCK_M;
    const int bh    = blockIdx.x / nQblk;
    const int qblk  = blockIdx.x % nQblk;
    const int b     = bh / Hc;

    const float* Qb = Q + (size_t)bh * Sc * Dc;
    const float* Kb = K + (size_t)bh * Sc * Dc;
    const float* Vb = V + (size_t)bh * Sc * Dc;
    float*       Ob = O + (size_t)bh * Sc * Dc;
    const int*   Mb = Msk + (size_t)b * Sc;

    const int qr = qblk * BLOCK_M + wave * 16 + l16;
    v16h bQ[2];
    {
        const float* qp = Qb + (size_t)qr * Dc;
#pragma unroll
        for (int c = 0; c < 2; ++c)
#pragma unroll
            for (int e = 0; e < 16; ++e)
                bQ[c][e] = (_Float16)(qp[c * 32 + hf * 16 + e] * QSCALE);
    }

    v8f o[4], lsum, zro;
#pragma unroll
    for (int r = 0; r < 8; ++r) { lsum[r] = 0.0f; zro[r] = 0.0f; }
#pragma unroll
    for (int t = 0; t < 4; ++t)
#pragma unroll
        for (int r = 0; r < 8; ++r) o[t][r] = 0.0f;
    float mst = -INFINITY;

    for (int kb = 0; kb < Sc; kb += BLOCK_N) {
        __syncthreads();
        {   // K staging (convert in-kernel)
            const int r  = tid >> 1;
            const int hd = (tid & 1) * 32;
            const float* kp = Kb + (size_t)(kb + r) * Dc + hd;
#pragma unroll
            for (int g = 0; g < 4; ++g) {
                v8h h;
#pragma unroll
                for (int j = 0; j < 4; ++j) {
                    v2h pk = cvt_pk(kp[g*8 + 2*j], kp[g*8 + 2*j + 1]);
                    h[2*j] = pk[0]; h[2*j+1] = pk[1];
                }
                *(v8h*)&Kl[r * 64 + hd + g * 8] = h;
            }
        }
        {   // V staging: transpose + mask-zero
            const int kv0 = (tid & 15) * 4;
            const int d0  = (tid >> 4) * 8;
            float kf[4];
#pragma unroll
            for (int i = 0; i < 4; ++i) kf[i] = Mb[kb + kv0 + i] ? 1.0f : 0.0f;
#pragma unroll
            for (int j = 0; j < 8; ++j) {
                v4h h;
#pragma unroll
                for (int i = 0; i < 4; ++i)
                    h[i] = (_Float16)(Vb[(size_t)(kb + kv0 + i) * Dc + d0 + j] * kf[i]);
                *(v4h*)&VTl[(d0 + j) * 64 + kv0] = h;
            }
            if (tid < BLOCK_N) keepL[tid] = (_Float16)(Mb[kb + tid] ? 1.0f : 0.0f);
        }
        __syncthreads();

        fa_block(Kl, VTl, keepL, l16, hf, bQ, o, lsum, mst, zro);
    }

    fa_epilogue(Ob, qr, hf, o, lsum);
}

extern "C" void kernel_launch(void* const* d_in, const int* in_sizes, int n_in,
                              void* d_out, int out_size, void* d_ws, size_t ws_size,
                              hipStream_t stream) {
    (void)in_sizes; (void)n_in; (void)out_size;
    const float* Q  = (const float*)d_in[0];
    const float* K  = (const float*)d_in[1];
    const float* V  = (const float*)d_in[2];
    const int*   Mk = (const int*)d_in[3];
    float* out = (float*)d_out;

    const size_t elems = (size_t)NBH * Sc * Dc;             // 4,194,304
    const size_t need  = elems * 2 * sizeof(_Float16) + (size_t)Bc * Sc * sizeof(_Float16);

    dim3 grid(NBH * (Sc / BLOCK_M));   // 1024 workgroups
    dim3 block(128);                   // 4 wave32

    if (ws_size >= need) {
        _Float16* Kh    = (_Float16*)d_ws;
        _Float16* VTh   = Kh + elems;
        _Float16* keepH = VTh + elems;
        conv_k_kernel <<<dim3(elems / (256 * 8)), dim3(256), 0, stream>>>(K, Kh);
        conv_vt_kernel<<<dim3(elems / (256 * 8)), dim3(256), 0, stream>>>(V, Mk, VTh, keepH);
        fa_fwd_pre<<<grid, block, 0, stream>>>(Q, Kh, VTh, keepH, out);
    } else {
        fa_fwd_kernel<<<grid, block, 0, stream>>>(Q, K, V, Mk, out);
    }
}
